// DacResidualVectorQuantize_49228915147005
// MI455X (gfx1250) — compile-verified, matrix operand order checked by reading, not run on hardware
//
#include <hip/hip_runtime.h>
#include <hip/hip_bf16.h>

typedef __attribute__((ext_vector_type(16))) __bf16 v16bf;
typedef __attribute__((ext_vector_type(8)))  float  v8f;
typedef __attribute__((ext_vector_type(4)))  float  v4f;

#define NCB   9
#define HID   1024
#define CBD   8
#define CBS   1024
#define TLEN  8192
// col stride 1044 floats: 16B-aligned, col->bank shift = 20 banks (gcd 4 with 64)
// -> the 16 lanes of a half-wave hit 16 disjoint 4-bank windows on ds_load_b128.
#define RSTRIDE 1044

// ---- async global->LDS copy availability (gfx1250 GLOBAL_LOAD_ASYNC_TO_LDS_*) ----
#if defined(__has_builtin)
# if __has_builtin(__builtin_amdgcn_global_load_async_to_lds_b32)
#  define HAVE_ASYNC_COPY 1
# endif
#endif
#ifndef HAVE_ASYNC_COPY
# define HAVE_ASYNC_COPY 0
#endif
#if HAVE_ASYNC_COPY
typedef __attribute__((address_space(1))) int g1_int;   // global (AS1) int*
typedef __attribute__((address_space(3))) int l3_int;   // LDS (AS3) int*
#endif

// ---- workspace layout (bytes) ----
#define WS_LOSS 0
#define WS_WA   64                               // in_proj A-frags:  9*32ksteps*32lanes*16 bf16
#define WA_ELE  (NCB*32*32*16)
#define WS_CB   (WS_WA + WA_ELE*2)               // dist B-frags (normalized cb): 9*64jtiles*32*16
#define CB_ELE  (NCB*64*32*16)
#define WS_WO   (WS_CB + CB_ELE*2)               // out_proj A-frags: 9*64mtiles*32*16
#define WO_ELE  (NCB*64*32*16)

// ---- output layout (floats) ----
#define OUT_Q    0
#define OUT_CODE ((size_t)4*HID*TLEN)                    // 33554432
#define OUT_LAT  (OUT_CODE + (size_t)4*NCB*TLEN)         // 33849344
#define OUT_CL   (OUT_LAT + (size_t)4*NCB*CBD*TLEN)      // 36208640
#define OUT_CB   (OUT_CL + 4)

// Pre-swizzle weights into CDNA5 WMMA bf16 fragment layout (16-bit A 16x32):
//  lane<16 : M=lane,    slots 0..7 -> K {0..7},   slots 8..15 -> K {16..23}
//  lane>=16: M=lane-16, slots 0..7 -> K {8..15},  slots 8..15 -> K {24..31}
__global__ void rvq_prep(const float* __restrict__ w_in,
                         const float* __restrict__ cb,
                         const float* __restrict__ w_out,
                         void* ws) {
  int tid = blockIdx.x * 256 + threadIdx.x;
  if (tid == 0) *(float*)((char*)ws + WS_LOSS) = 0.f;
  __bf16* wa  = (__bf16*)((char*)ws + WS_WA);
  __bf16* cbf = (__bf16*)((char*)ws + WS_CB);
  __bf16* wo  = (__bf16*)((char*)ws + WS_WO);

  if (tid < NCB*32*32) {                       // in_proj A: W_in is (8 x 1024), rows 8..15 zero-pad
    int l = tid & 31, kk = (tid >> 5) & 31, i = tid >> 10;
    __bf16* dst = wa + (size_t)tid * 16;
    int M = l & 15;
    int khalf = (l < 16) ? 0 : 8;
#pragma unroll
    for (int s = 0; s < 16; s++) {
      int k = kk * 32 + ((s < 8) ? s : s + 8) + khalf;
      float v = (M < 8) ? w_in[(size_t)(i * 8 + M) * HID + k] : 0.f;
      dst[s] = (__bf16)v;
    }
  } else if (tid < NCB*32*32 + NCB*64*32) {    // dist B (K=8 pad to 32): normalized codebook rows
    int r = tid - NCB*32*32;
    int l = r & 31, jt = (r >> 5) & 63, i = r >> 11;
    __bf16* dst = cbf + (size_t)r * 16;
    float vals[16];
#pragma unroll
    for (int s = 0; s < 16; s++) vals[s] = 0.f;
    if (l < 16) {                              // lanes>=16 carry K 8..15 / 24..31 -> zero pad
      int j = jt * 16 + l;
      const float* row = cb + ((size_t)i * CBS + j) * CBD;
      float t[8], n2 = 0.f;
#pragma unroll
      for (int s = 0; s < 8; s++) { t[s] = row[s]; n2 += t[s] * t[s]; }
      float inv = 1.f / fmaxf(sqrtf(n2), 1e-12f);
#pragma unroll
      for (int s = 0; s < 8; s++) vals[s] = t[s] * inv;
    }
#pragma unroll
    for (int s = 0; s < 16; s++) dst[s] = (__bf16)vals[s];
  } else if (tid < NCB*32*32 + 2*NCB*64*32) {  // out_proj A (K=8 pad to 32)
    int r = tid - (NCB*32*32 + NCB*64*32);
    int l = r & 31, mt = (r >> 5) & 63, i = r >> 11;
    __bf16* dst = wo + (size_t)r * 16;
    float vals[16];
#pragma unroll
    for (int s = 0; s < 16; s++) vals[s] = 0.f;
    if (l < 16) {
      int m = mt * 16 + l;
      const float* row = w_out + ((size_t)i * HID + m) * CBD;
#pragma unroll
      for (int s = 0; s < 8; s++) vals[s] = row[s];
    }
#pragma unroll
    for (int s = 0; s < 16; s++) dst[s] = (__bf16)vals[s];
  }
}

// One wave owns a 16-column tile; residual (1024 x 16, f32) lives in LDS.
// 4 waves / workgroup -> 261KB of the WGP's 320KB LDS.
__global__ __launch_bounds__(128)
void rvq_main(const float* __restrict__ hidden,
              const float* __restrict__ b_in,
              const float* __restrict__ b_out,
              const float* __restrict__ codebooks,
              float* __restrict__ out,
              void* ws) {
  extern __shared__ float lds[];
  const int lane = threadIdx.x & 31;
  const int warp = threadIdx.x >> 5;
  const int tile = blockIdx.x * 4 + warp;
  const int b  = tile >> 9;                 // 512 tiles of 16 columns per batch
  const int t0 = (tile & 511) * 16;
  const int col = lane & 15;
  const int hi  = lane >> 4;                // which half of the wave

  float* R = lds + (size_t)warp * (RSTRIDE * 16);

  const __bf16* wa  = (const __bf16*)((const char*)ws + WS_WA);
  const __bf16* cbf = (const __bf16*)((const char*)ws + WS_CB);
  const __bf16* wo  = (const __bf16*)((const char*)ws + WS_WO);

  float* qout  = out + OUT_Q;
  float* codes = out + OUT_CODE;
  float* lat   = out + OUT_LAT;

  // ---- load residual tile transposed into LDS: R[col][channel] ----
#if HAVE_ASYNC_COPY
  {
    // ASYNCcnt-tracked scatter: per-lane LDS destination does the transpose directly.
    const float* gbase = hidden + (size_t)b * HID * TLEN + t0 + col;
#pragma unroll 4
    for (int cc = 0; cc < 512; ++cc) {
      int c = cc * 2 + hi;
      __builtin_amdgcn_global_load_async_to_lds_b32(
          (g1_int*)(gbase + (size_t)c * TLEN),
          (l3_int*)&R[col * RSTRIDE + c], 0, 0);
    }
# if __has_builtin(__builtin_amdgcn_s_wait_asynccnt)
    __builtin_amdgcn_s_wait_asynccnt(0);
# else
    asm volatile("s_wait_asynccnt 0x0" ::: "memory");
# endif
  }
#else
  // fallback: b128 global loads (4 consecutive t per lane), scalar LDS scatter
#pragma unroll 2
  for (int cc = 0; cc < 128; ++cc) {
    int c  = cc * 8 + (lane >> 2);
    int tq = (lane & 3) * 4;
    v4f h = *(const v4f*)(hidden + ((size_t)b * HID + c) * TLEN + t0 + tq);
#pragma unroll
    for (int j = 0; j < 4; j++) R[(tq + j) * RSTRIDE + c] = h[j];
  }
#endif
  __syncthreads();

  float loss = 0.f;

  for (int i = 0; i < NCB; i++) {
    // ================= in_proj: z(8 x 16) = W_in * R, K = 1024 =================
    v8f zc = {0.f,0.f,0.f,0.f,0.f,0.f,0.f,0.f};
    const __bf16* wai = wa + (size_t)i * 32 * 32 * 16;
    for (int kk = 0; kk < 32; kk++) {
      const __bf16* ap = wai + (size_t)(kk * 32 + lane) * 16;
      __builtin_prefetch(ap + 32 * 16, 0, 0);
      v16bf a = *(const v16bf*)ap;
      // B frag from LDS residual (f32 -> bf16 on the fly)
      const float* rc = R + col * RSTRIDE + kk * 32 + hi * 8;
      v4f x0 = *(const v4f*)(rc);
      v4f x1 = *(const v4f*)(rc + 4);
      v4f x2 = *(const v4f*)(rc + 16);
      v4f x3 = *(const v4f*)(rc + 20);
      v16bf bb;
#pragma unroll
      for (int s = 0; s < 4; s++) {
        bb[s]      = (__bf16)x0[s];
        bb[s + 4]  = (__bf16)x1[s];
        bb[s + 8]  = (__bf16)x2[s];
        bb[s + 12] = (__bf16)x3[s];
      }
      zc = __builtin_amdgcn_wmma_f32_16x16x32_bf16(false, a, false, bb,
                                                   (short)0, zc, false, false);
    }

    // lane<16, VGPR r holds z[dim r][col=lane]; add bias, emit latents
    float z[8];
#pragma unroll
    for (int r = 0; r < 8; r++) z[r] = 0.f;
    if (lane < 16) {
#pragma unroll
      for (int r = 0; r < 8; r++) {
        z[r] = zc[r] + b_in[i * CBD + r];
        lat[((size_t)b * (NCB * CBD) + i * CBD + r) * TLEN + t0 + col] = z[r];
      }
    }
    // L2 normalize (per column, 8-dim)
    float n2 = 0.f;
#pragma unroll
    for (int r = 0; r < 8; r++) n2 += z[r] * z[r];
    float inv = 1.f / fmaxf(sqrtf(n2), 1e-12f);
    v16bf a2;
#pragma unroll
    for (int s = 0; s < 16; s++) a2[s] = (__bf16)0.f;
    if (lane < 16) {
#pragma unroll
      for (int r = 0; r < 8; r++) a2[r] = (__bf16)(z[r] * inv);
    }

    // ================= dist: (16 cols) x (1024 codes), argmax cosine =================
    // dist = 2*dot + ||cb_n||^2 - l2(col); last two ~constant -> argmax(dot)
    const __bf16* cbi = cbf + (size_t)i * 64 * 32 * 16;
    float best[8];
    int   bidx[8];
#pragma unroll
    for (int r = 0; r < 8; r++) { best[r] = -3.4e38f; bidx[r] = 0; }
    for (int jt = 0; jt < 64; jt++) {
      const __bf16* bp = cbi + (size_t)(jt * 32 + lane) * 16;
      __builtin_prefetch(bp + 32 * 16, 0, 0);
      v16bf bcb = *(const v16bf*)bp;
      v8f d = {0.f,0.f,0.f,0.f,0.f,0.f,0.f,0.f};
      d = __builtin_amdgcn_wmma_f32_16x16x32_bf16(false, a2, false, bcb,
                                                  (short)0, d, false, false);
      int j = jt * 16 + col;                  // VGPR r -> col r (lanes<16) / col 8+r
#pragma unroll
      for (int r = 0; r < 8; r++) {
        if (d[r] > best[r]) { best[r] = d[r]; bidx[r] = j; }
      }
    }
    // reduce argmax across the 16 lanes of each half (xor 1,2,4,8 stays in half)
#pragma unroll
    for (int off = 1; off < 16; off <<= 1) {
#pragma unroll
      for (int r = 0; r < 8; r++) {
        float om = __shfl_xor(best[r], off, 32);
        int   oi = __shfl_xor(bidx[r], off, 32);
        if (om > best[r] || (om == best[r] && oi < bidx[r])) { best[r] = om; bidx[r] = oi; }
      }
    }
    // broadcast so lane l gets the index for col = l&15
    int colidx[16];
#pragma unroll
    for (int r = 0; r < 8; r++) {
      colidx[r]     = __shfl(bidx[r], 0, 32);
      colidx[r + 8] = __shfl(bidx[r], 16, 32);
    }
    int myidx = colidx[0];
#pragma unroll
    for (int s = 1; s < 16; s++) if (col == s) myidx = colidx[s];

    // codes, gather q from ORIGINAL codebook, loss, build out_proj B frag
    v16bf b3;
#pragma unroll
    for (int s = 0; s < 16; s++) b3[s] = (__bf16)0.f;
    if (lane < 16) {
      codes[((size_t)b * NCB + i) * TLEN + t0 + col] = (float)myidx;
      const float* qr = codebooks + ((size_t)i * CBS + myidx) * CBD;
      v4f q0 = *(const v4f*)qr;
      v4f q1 = *(const v4f*)(qr + 4);
#pragma unroll
      for (int r = 0; r < 4; r++) {
        float d0 = z[r] - q0[r], d1 = z[r + 4] - q1[r];
        loss += d0 * d0 + d1 * d1;
        b3[r]     = (__bf16)q0[r];
        b3[r + 4] = (__bf16)q1[r];
      }
    }

    // ================= out_proj: O(1024 x 16) = W_out * q; residual -= O + b =================
    const __bf16* woi = wo + (size_t)i * 64 * 32 * 16;
    for (int mt = 0; mt < 64; mt++) {
      const __bf16* ap = woi + (size_t)(mt * 32 + lane) * 16;
      __builtin_prefetch(ap + 32 * 16, 0, 0);
      v16bf a3 = *(const v16bf*)ap;
      v8f o = {0.f,0.f,0.f,0.f,0.f,0.f,0.f,0.f};
      o = __builtin_amdgcn_wmma_f32_16x16x32_bf16(false, a3, false, b3,
                                                  (short)0, o, false, false);
      int m0 = mt * 16 + hi * 8;              // lanes<16 rows m0..m0+7, lanes>=16 rows m0+8..15
      const float* bo = b_out + (size_t)i * HID + m0;
      v4f bo0 = *(const v4f*)bo;
      v4f bo1 = *(const v4f*)(bo + 4);
      float* rp = R + col * RSTRIDE + m0;
      v4f r0 = *(const v4f*)rp;
      v4f r1 = *(const v4f*)(rp + 4);
#pragma unroll
      for (int r = 0; r < 4; r++) {
        r0[r] -= o[r] + bo0[r];
        r1[r] -= o[r + 4] + bo1[r];
      }
      *(v4f*)rp = r0;
      *(v4f*)(rp + 4) = r1;
    }
    __syncthreads();   // cheap 4-wave barrier; guarantees LDS RAW ordering before next codebook
  }

  // ---- quantized = hidden - final residual (sum of outs telescopes); b128 streams ----
#pragma unroll 2
  for (int cc = 0; cc < 128; ++cc) {
    int c  = cc * 8 + (lane >> 2);
    int tq = (lane & 3) * 4;
    size_t g = ((size_t)b * HID + c) * TLEN + t0 + tq;
    v4f h = *(const v4f*)(hidden + g);
    v4f r;
#pragma unroll
    for (int j = 0; j < 4; j++) r[j] = h[j] - R[(tq + j) * RSTRIDE + c];
    *(v4f*)(qout + g) = r;
  }

  // loss: commit == cb numerically; reduce wave then one atomic
#pragma unroll
  for (int off = 16; off >= 1; off >>= 1) loss += __shfl_xor(loss, off, 32);
  if (lane == 0) atomicAdd((float*)((char*)ws + WS_LOSS), loss);
}

__global__ void rvq_final(const void* ws, float* out) {
  if (threadIdx.x == 0 && blockIdx.x == 0) {
    // sum_i mean_i over (B,8,T) = total / (4*8*8192)
    float v = *(const float*)((const char*)ws + WS_LOSS) / 262144.f;
#pragma unroll
    for (int bb = 0; bb < 4; bb++) {
      out[OUT_CL + bb] = v;
      out[OUT_CB + bb] = v;
    }
  }
}

extern "C" void kernel_launch(void* const* d_in, const int* in_sizes, int n_in,
                              void* d_out, int out_size, void* d_ws, size_t ws_size,
                              hipStream_t stream) {
  const float* hidden = (const float*)d_in[0];
  const float* wi     = (const float*)d_in[1];
  const float* bi     = (const float*)d_in[2];
  const float* wo     = (const float*)d_in[3];
  const float* bo     = (const float*)d_in[4];
  const float* cb     = (const float*)d_in[5];
  (void)in_sizes; (void)n_in; (void)out_size; (void)ws_size;

  // 1) swizzle weights/codebooks into WMMA fragment tables in ws (+ zero loss acc)
  rvq_prep<<<180, 256, 0, stream>>>(wi, cb, wo, d_ws);

  // 2) main fused RVQ: 2048 column tiles, 4 waves/WG, 261KB dynamic LDS per WG
  size_t smem = (size_t)RSTRIDE * 16 * 4 * sizeof(float);
  rvq_main<<<512, 128, smem, stream>>>(hidden, bi, bo, cb, (float*)d_out, d_ws);

  // 3) finalize losses
  rvq_final<<<1, 1, 0, stream>>>(d_ws, (float*)d_out);
}
